// Decoder_41360535060514
// MI455X (gfx1250) — matrix operand order repaired and checked by reading, not compiled, not run on hardware
//
#include <hip/hip_runtime.h>

#define D_FEAT 128
#define E_POS_C 262144
#define E_NEG_C 262144
#define E_TOT (E_POS_C + E_NEG_C)
#define TILE_M 32                   // edges per block iteration (2 x 16-row WMMA subtiles)
#define N_TILES (E_TOT / TILE_M)    // 16384 full tiles
#define ASTR 264                    // A row stride in halves (256+8): 528B, conflict-free frags
#define HSTR 132                    // H row stride in floats (128+4): 528B, conflict-free stores

typedef __attribute__((ext_vector_type(16))) __bf16          v16bf;
typedef __attribute__((ext_vector_type(8)))  float           v8f;
typedef __attribute__((ext_vector_type(16))) unsigned short  v16us;
typedef __attribute__((ext_vector_type(8)))  unsigned short  v8us;

// fp32 -> bf16 round-to-nearest-even (scalar fallback)
static __device__ __forceinline__ unsigned short f2bf(float f) {
  unsigned int u = __builtin_bit_cast(unsigned int, f);
  u += 0x7FFFu + ((u >> 16) & 1u);
  return (unsigned short)(u >> 16);
}

// pack two fp32 -> packed bf16x2 (one v_cvt_pk_bf16_f32 when available)
static __device__ __forceinline__ unsigned int pk2bf(float f0, float f1) {
#if defined(__has_builtin) && __has_builtin(__builtin_amdgcn_cvt_pk_bf16_f32)
  typedef __attribute__((ext_vector_type(2))) __bf16 v2bf;
  v2bf v = __builtin_amdgcn_cvt_pk_bf16_f32(f0, f1);
  return __builtin_bit_cast(unsigned int, v);
#else
  return (unsigned)f2bf(f0) | ((unsigned)f2bf(f1) << 16);
#endif
}

// Build one 16x32 bf16 A fragment from LDS (two 16B loads per lane).
static __device__ __forceinline__ v16bf loadA(const unsigned short* ap, int kb) {
  v8us lo  = *(const v8us*)(ap + kb * 32);        // K chunk + {0..7}  (or {8..15})
  v8us hi8 = *(const v8us*)(ap + kb * 32 + 16);   // K chunk + {16..23} (or {24..31})
  v16us au = __builtin_shufflevector(lo, hi8,
                0,1,2,3,4,5,6,7, 8,9,10,11,12,13,14,15);
  return __builtin_bit_cast(v16bf, au);
}

__global__ __launch_bounds__(256) void edge_mlp_wmma(
    const float* __restrict__ x,
    const long long* __restrict__ posE,   // [2, E_POS]
    const long long* __restrict__ negE,   // [2, E_NEG]
    const float* __restrict__ W1,         // [256, 128] row-major
    const float* __restrict__ b1,         // [128]
    const float* __restrict__ W2,         // [128, 1]
    const float* __restrict__ b2,         // [1]
    float* __restrict__ out)              // [E, 1]
{
  // A (32x264 bf16) and H (32x132 f32) are both 16896 bytes -> one buffer.
  __shared__ __align__(16) unsigned char smem[TILE_M * ASTR * 2];
  unsigned short* Alds = (unsigned short*)smem;
  float*          Hlds = (float*)smem;

  const int tid  = threadIdx.x;
  const int lane = tid & 31;
  const int wave = tid >> 5;                    // 0..7, each owns 16 output cols
  const int col  = (wave << 4) + (lane & 15);   // output feature column 0..127
  const int hi   = lane >> 4;                   // half-wave select

  // ---- B fragments: 16 W1 columns per wave, bf16, loaded once ------------
  v16bf bfrag[8];
  {
    const int koff = hi ? 8 : 0;
#pragma unroll
    for (int kb = 0; kb < 8; ++kb) {
      unsigned int t[8];
#pragma unroll
      for (int k = 0; k < 8; ++k)
        t[k] = pk2bf(W1[(kb * 32 + koff + 2 * k + (k >= 4 ? 8 : 0)) * D_FEAT + col],
                     W1[(kb * 32 + koff + 2 * k + (k >= 4 ? 8 : 0) + 1) * D_FEAT + col]);
      v16us u;
#pragma unroll
      for (int k = 0; k < 8; ++k) {
        u[2 * k]     = (unsigned short)(t[k] & 0xFFFFu);
        u[2 * k + 1] = (unsigned short)(t[k] >> 16);
      }
      bfrag[kb] = __builtin_bit_cast(v16bf, u);
    }
  }
  const float b1c = b1[col];
  const float b2v = b2[0];

  // layer-2 assignment: thread t -> edge m2 = t>>3, 16-col chunk ch = t&7
  const int m2 = tid >> 3;
  const int ch = tid & 7;
  float w2reg[16];
#pragma unroll
  for (int k = 0; k < 16; ++k) w2reg[k] = W2[ch * 16 + k];

  // staging assignment: 32 rows, 8 threads per row, 8 float4 chunks each
  const int srow = tid >> 3;        // 0..31 (edge row)
  const int schk = tid & 7;         // float4 chunk phase within row

  for (int tile = blockIdx.x; tile < N_TILES; tile += gridDim.x) {
    const int base = tile * TILE_M;

    __syncthreads();                  // prev tile's H fully consumed

    // ---- stage A tile: rows = edges, cols 0..127 = x[src], 128..255 = x[tar]
    {
      const int e = base + srow;
      long long s, t;
      if (e < E_POS_C) { s = posE[e];           t = posE[E_POS_C + e]; }
      else             { s = negE[e - E_POS_C]; t = negE[E_NEG_C + (e - E_POS_C)]; }
      const float* xs = x + s * D_FEAT;
      const float* xt = x + t * D_FEAT;
      unsigned short* arow = &Alds[srow * ASTR];
#pragma unroll
      for (int j = 0; j < 8; ++j) {
        const int c4 = (schk + j * 8) * 4;                 // 0..252, step 32 (+phase)
        const float4 v = (c4 < D_FEAT) ? *(const float4*)(xs + c4)
                                       : *(const float4*)(xt + c4 - D_FEAT);
        uint2 p;
        p.x = pk2bf(v.x, v.y);
        p.y = pk2bf(v.z, v.w);
        *(uint2*)(arow + c4) = p;                          // ds_store_b64
      }
    }
    __syncthreads();

    // ---- GEMM: two 16(M)x16(N) tiles per wave, K = 256 in 8 chunks -------
    v8f acc0 = {}, acc1 = {};
    {
      const int koffA = hi ? 8 : 0;      // lanes>=16 take K {8..15,24..31}
      const unsigned short* ap0 = &Alds[(lane & 15) * ASTR + koffA];
      const unsigned short* ap1 = ap0 + 16 * ASTR;
      v16bf a0 = loadA(ap0, 0);
      v16bf a1 = loadA(ap1, 0);
#pragma unroll
      for (int kb = 0; kb < 8; ++kb) {
        acc0 = __builtin_amdgcn_wmma_f32_16x16x32_bf16(
                   false, a0, false, bfrag[kb], (short)0, acc0, false, false);
        acc1 = __builtin_amdgcn_wmma_f32_16x16x32_bf16(
                   false, a1, false, bfrag[kb], (short)0, acc1, false, false);
        if (kb < 7) {                     // pipeline next A fragments
          a0 = loadA(ap0, kb + 1);
          a1 = loadA(ap1, kb + 1);
        }
      }
    }
    __syncthreads();                  // all waves done reading A -> reuse as H

    // ---- store H = C + b1 (row-major, conflict-free: +8 rows = +32 banks) -
    {
      const int mh = hi ? 8 : 0;
#pragma unroll
      for (int r = 0; r < 8; ++r) {
        Hlds[(r + mh) * HSTR + col]      = acc0[r] + b1c;   // ds_store_b32
        Hlds[(16 + r + mh) * HSTR + col] = acc1[r] + b1c;
      }
    }
    __syncthreads();

    // ---- layer 2: out[m] = relu(H[m]) . W2 + b2 ---------------------------
    {
      const float* hp = &Hlds[m2 * HSTR + ch * 16];
      float4 h0 = *(const float4*)(hp + 0);
      float4 h1 = *(const float4*)(hp + 4);
      float4 h2 = *(const float4*)(hp + 8);
      float4 h3 = *(const float4*)(hp + 12);
      float p = 0.0f;
      p = fmaf(fmaxf(h0.x, 0.0f), w2reg[0],  p);
      p = fmaf(fmaxf(h0.y, 0.0f), w2reg[1],  p);
      p = fmaf(fmaxf(h0.z, 0.0f), w2reg[2],  p);
      p = fmaf(fmaxf(h0.w, 0.0f), w2reg[3],  p);
      p = fmaf(fmaxf(h1.x, 0.0f), w2reg[4],  p);
      p = fmaf(fmaxf(h1.y, 0.0f), w2reg[5],  p);
      p = fmaf(fmaxf(h1.z, 0.0f), w2reg[6],  p);
      p = fmaf(fmaxf(h1.w, 0.0f), w2reg[7],  p);
      p = fmaf(fmaxf(h2.x, 0.0f), w2reg[8],  p);
      p = fmaf(fmaxf(h2.y, 0.0f), w2reg[9],  p);
      p = fmaf(fmaxf(h2.z, 0.0f), w2reg[10], p);
      p = fmaf(fmaxf(h2.w, 0.0f), w2reg[11], p);
      p = fmaf(fmaxf(h3.x, 0.0f), w2reg[12], p);
      p = fmaf(fmaxf(h3.y, 0.0f), w2reg[13], p);
      p = fmaf(fmaxf(h3.z, 0.0f), w2reg[14], p);
      p = fmaf(fmaxf(h3.w, 0.0f), w2reg[15], p);
      // reduce the 8 chunk-partials of this edge (threads t..t+7, same wave)
      p += __shfl_xor(p, 1, 8);
      p += __shfl_xor(p, 2, 8);
      p += __shfl_xor(p, 4, 8);
      if (ch == 0) out[base + m2] = p + b2v;
    }
  }
}

extern "C" void kernel_launch(void* const* d_in, const int* in_sizes, int n_in,
                              void* d_out, int out_size, void* d_ws, size_t ws_size,
                              hipStream_t stream) {
  (void)in_sizes; (void)n_in; (void)out_size; (void)d_ws; (void)ws_size;
  const float*     x    = (const float*)d_in[0];
  const long long* posE = (const long long*)d_in[1];
  const long long* negE = (const long long*)d_in[2];
  const float*     W1   = (const float*)d_in[3];
  const float*     b1   = (const float*)d_in[4];
  const float*     W2   = (const float*)d_in[5];
  const float*     b2   = (const float*)d_in[6];
  float*           out  = (float*)d_out;

  edge_mlp_wmma<<<2048, 256, 0, stream>>>(x, posE, negE, W1, b1, W2, b2, out);
}
